// LESSR_90091234001300
// MI455X (gfx1250) — matrix operand order repaired and checked by reading, not compiled, not run on hardware
//
#include <hip/hip_runtime.h>
#include <hip/hip_bf16.h>
#include <math.h>

// Problem sizes (fixed by the reference)
#define V_ 50000
#define D_ 128
#define N_ 20480
#define B_ 2048
#define E_ 81920
#define T_ 4

typedef __attribute__((ext_vector_type(16))) __bf16 v16bf;
typedef __attribute__((ext_vector_type(8)))  float  v8f;

__device__ __forceinline__ unsigned short f2bf(float f) {
  unsigned u = __float_as_uint(f);
  u += 0x7FFFu + ((u >> 16) & 1u);       // round-to-nearest-even
  return (unsigned short)(u >> 16);
}
__device__ __forceinline__ float sigm(float x) { return 1.f / (1.f + __expf(-x)); }
// order-preserving float<->uint encode for atomicMax-based segment max
__device__ __forceinline__ unsigned fenc(float f) {
  unsigned u = __float_as_uint(f);
  return (u >> 31) ? ~u : (u | 0x80000000u);
}
__device__ __forceinline__ float fdec(unsigned e) {
  unsigned u = (e & 0x80000000u) ? (e & 0x7FFFFFFFu) : ~e;
  return __uint_as_float(u);
}

// ---------------------------------------------------------------------------
// WMMA GEMM: C[M,Nc] = A[M,K](bf16,row-major, lda) x BT[Nc,K](bf16,row-major, ldb)
// flags bit0: add bias[col]; bit1: accumulate into existing C.
// Register-blocked 2x2: each wave computes a 32x32 output block (4 16x16 tiles)
// -> per K=32 step: 2 A-frags + 2 B-frags (8x b128 loads) feed 4 WMMAs.
// Per-lane operand layout (CDNA5 ISA 7.12.2, 16-bit A / B):
//   A: lane m=l&15, kh=l>>4; elems {k0+8kh+0..7, k0+8kh+16..23} -> two b128 loads
//   B^T row n=l&15, kh=l>>4; elems {k0+8kh+0..7, k0+8kh+16..23} -> two b128 loads
//   D: elem r -> row = 16*tm + r + 8*kh, col = 16*tn + (l&15)
// M must be a multiple of 32 (all call sites: 20480 / 2048). Nc multiple of 16;
// odd tile counts (scores GEMM: 3125) handled by a wave-uniform hasN1 branch,
// keeping EXEC all-1s around every WMMA.
// ---------------------------------------------------------------------------
union FragBF { uint4 q[2]; v16bf v; };

__device__ __forceinline__ v8f wmma_bf16(const FragBF& a, const FragBF& b, v8f c) {
  return __builtin_amdgcn_wmma_f32_16x16x32_bf16(
      /*neg_a=*/false, a.v, /*neg_b=*/false, b.v,
      /*c_mod=*/(short)0, c, /*reuse_a=*/false, /*reuse_b=*/false);
}

__device__ __forceinline__ void load_fragA(FragBF& f, const unsigned short* row,
                                           int k0, int kh) {
  f.q[0] = *(const uint4*)(row + k0 + kh * 8);
  f.q[1] = *(const uint4*)(row + k0 + kh * 8 + 16);
}
__device__ __forceinline__ void load_fragB(FragBF& f, const unsigned short* row,
                                           int k0, int kh) {
  f.q[0] = *(const uint4*)(row + k0 + kh * 8);
  f.q[1] = *(const uint4*)(row + k0 + kh * 8 + 16);
}

__device__ __forceinline__ void store_tile(const v8f& acc, float* __restrict__ C, int ldc,
                                           int rowBase, int col,
                                           const float* __restrict__ bias, int flags) {
  float bsv = (flags & 1) ? bias[col] : 0.f;
#pragma unroll
  for (int r = 0; r < 8; ++r) {
    size_t idx = (size_t)(rowBase + r) * ldc + col;
    float v = acc[r] + bsv;
    if (flags & 2) v += C[idx];
    C[idx] = v;
  }
}

__global__ void __launch_bounds__(128)
k_gemm_bf16(const unsigned short* __restrict__ A, int lda,
            const unsigned short* __restrict__ BT, int ldb,
            float* __restrict__ C, int ldc,
            int M, int Ncols, int K,
            const float* __restrict__ bias, int flags)
{
  const int wave = threadIdx.x >> 5;
  const int lane = threadIdx.x & 31;
  const int tilesN = Ncols >> 4;
  const int tn0 = (blockIdx.x * 4 + wave) * 2;
  const int tm0 = blockIdx.y * 2;
  if (tn0 >= tilesN) return;                   // wave-uniform
  const bool hasN1 = (tn0 + 1) < tilesN;       // wave-uniform
  const int m16 = lane & 15;
  const int kh  = lane >> 4;

  const unsigned short* a0 = A + (size_t)(tm0 * 16 + m16) * lda;
  const unsigned short* a1 = a0 + (size_t)16 * lda;
  const unsigned short* b0 = BT + (size_t)(tn0 * 16 + m16) * ldb;
  const unsigned short* b1 = b0 + (size_t)16 * ldb;

  v8f c00 = {}, c01 = {}, c10 = {}, c11 = {};
  if (hasN1) {
    for (int k0 = 0; k0 < K; k0 += 32) {
      FragBF fa0, fa1, fb0, fb1;
      load_fragA(fa0, a0, k0, kh);
      load_fragA(fa1, a1, k0, kh);
      load_fragB(fb0, b0, k0, kh);
      load_fragB(fb1, b1, k0, kh);
      __builtin_prefetch(b0 + k0 + 32 + kh * 16, 0, 0);
      __builtin_prefetch(b1 + k0 + 32 + kh * 16, 0, 0);
      c00 = wmma_bf16(fa0, fb0, c00);
      c10 = wmma_bf16(fa1, fb0, c10);
      c01 = wmma_bf16(fa0, fb1, c01);
      c11 = wmma_bf16(fa1, fb1, c11);
    }
  } else {
    for (int k0 = 0; k0 < K; k0 += 32) {
      FragBF fa0, fa1, fb0;
      load_fragA(fa0, a0, k0, kh);
      load_fragA(fa1, a1, k0, kh);
      load_fragB(fb0, b0, k0, kh);
      __builtin_prefetch(b0 + k0 + 32 + kh * 16, 0, 0);
      c00 = wmma_bf16(fa0, fb0, c00);
      c10 = wmma_bf16(fa1, fb0, c10);
    }
  }

  const int row0 = tm0 * 16 + 8 * kh;
  const int col0 = tn0 * 16 + m16;
  store_tile(c00, C, ldc, row0, col0, bias, flags);
  store_tile(c10, C, ldc, row0 + 16, col0, bias, flags);
  if (hasN1) {
    store_tile(c01, C, ldc, row0, col0 + 16, bias, flags);
    store_tile(c11, C, ldc, row0 + 16, col0 + 16, bias, flags);
  }
}

// ---------------------------------------------------------------------------
// Elementwise / reduction helper kernels
// ---------------------------------------------------------------------------
__global__ void k_norm_emb(const float* __restrict__ emb,
                           float* __restrict__ tn, unsigned short* __restrict__ tb) {
  int r = blockIdx.x, d = threadIdx.x;
  float v = emb[(size_t)r * D_ + d];
  __shared__ float sm[128];
  sm[d] = v * v; __syncthreads();
  for (int o = 64; o > 0; o >>= 1) { if (d < o) sm[d] += sm[d + o]; __syncthreads(); }
  float nrm = sqrtf(sm[0]);
  float s = nrm > 1.f ? 1.f / nrm : 1.f;     // max_norm = 1
  float o = v * s;
  tn[(size_t)r * D_ + d] = o;
  tb[(size_t)r * D_ + d] = f2bf(o);
}

__global__ void k_convertT(const float* __restrict__ W, unsigned short* __restrict__ WT,
                           int K, int Nc) {
  int i = blockIdx.x * blockDim.x + threadIdx.x;
  if (i >= K * Nc) return;
  int k = i / Nc, n = i - k * Nc;
  WT[(size_t)n * K + k] = f2bf(W[i]);        // transpose + bf16
}

__global__ void k_bn_stats(const float* __restrict__ x, int stride, int nrows,
                           float* __restrict__ mu, float* __restrict__ rs) {
  int c = blockIdx.x, t = threadIdx.x;
  float s = 0.f, s2 = 0.f;
  for (int r = t; r < nrows; r += blockDim.x) {
    float v = x[(size_t)r * stride + c];
    s += v; s2 += v * v;
  }
  __shared__ float sh[256], sh2[256];
  sh[t] = s; sh2[t] = s2; __syncthreads();
  for (int o = 128; o > 0; o >>= 1) {
    if (t < o) { sh[t] += sh[t + o]; sh2[t] += sh2[t + o]; }
    __syncthreads();
  }
  if (t == 0) {
    float m = sh[0] / nrows;
    float var = sh2[0] / nrows - m * m;      // biased var (torch BN)
    mu[c] = m; rs[c] = rsqrtf(var + 1e-5f);
  }
}

__global__ void k_bn_apply(const float* __restrict__ x, int stride,
                           const float* __restrict__ mu, const float* __restrict__ rs,
                           const float* __restrict__ g, const float* __restrict__ b,
                           float* __restrict__ of, unsigned short* __restrict__ ob,
                           int C, int rows) {
  int i = blockIdx.x * blockDim.x + threadIdx.x;
  if (i >= rows * C) return;
  int r = i / C, c = i - r * C;
  float v = x[(size_t)r * stride + c];
  float y = g[c] * (v - mu[c]) * rs[c] + b[c];
  of[i] = y; ob[i] = f2bf(y);
}

__global__ void k_gather_rows(const float* __restrict__ src, int srcStride,
                              const int* __restrict__ idx,
                              float* __restrict__ dstF, unsigned short* __restrict__ dstB,
                              int dstStride, int width, int rows) {
  int i = blockIdx.x * blockDim.x + threadIdx.x;
  if (i >= rows * width) return;
  int r = i / width, d = i - r * width;
  float v = src[(size_t)idx[r] * srcStride + d];
  dstF[(size_t)r * dstStride + d] = v;
  if (dstB) dstB[(size_t)r * dstStride + d] = f2bf(v);
}

__global__ void k_gru_step(const float* __restrict__ Gi, const float* __restrict__ Gh,
                           const int* __restrict__ mbidx, const unsigned char* __restrict__ mbmask,
                           int t, float* __restrict__ h, unsigned short* __restrict__ hb,
                           int din, int n) {
  int i = blockIdx.x * blockDim.x + threadIdx.x;
  if (i >= n * din) return;
  int nd = i / din, d = i - nd * din;
  int g = mbidx[nd * T_ + t];
  int tri = 3 * din;
  float ir = Gi[(size_t)g * tri + d];
  float iz = Gi[(size_t)g * tri + din + d];
  float in_ = Gi[(size_t)g * tri + 2 * din + d];
  float hr = Gh[(size_t)nd * tri + d];
  float hz = Gh[(size_t)nd * tri + din + d];
  float hn = Gh[(size_t)nd * tri + 2 * din + d];
  float r = sigm(ir + hr);
  float z = sigm(iz + hz);
  float nn = tanhf(in_ + r * hn);
  float hold = h[i];
  float hnew = (1.f - z) * nn + z * hold;
  float out = mbmask[nd * T_ + t] ? hnew : hold;
  h[i] = out; hb[i] = f2bf(out);
}

__global__ void k_prelu_copy(const float* __restrict__ src, int ss,
                             float* __restrict__ dst, int ds,
                             const float* __restrict__ a, int rows) {
  int i = blockIdx.x * blockDim.x + threadIdx.x;
  if (i >= rows * 128) return;
  int r = i >> 7, d = i & 127;
  float v = src[(size_t)r * ss + d];
  dst[(size_t)r * ds + d] = v > 0.f ? v : a[d] * v;
}

__global__ void k_edge_score(const float* __restrict__ q, const float* __restrict__ k,
                             const float* __restrict__ We,
                             const int* __restrict__ src, const int* __restrict__ dst,
                             float* __restrict__ e, int ne) {
  int i = blockIdx.x * blockDim.x + threadIdx.x;
  if (i >= ne) return;
  const float* qp = q + (size_t)src[i] * 128;
  const float* kp = k + (size_t)dst[i] * 128;
  float acc = 0.f;
#pragma unroll 8
  for (int d = 0; d < 128; ++d) acc += We[d] * sigm(qp[d] + kp[d]);
  e[i] = acc;
}

__global__ void k_node_score(const float* __restrict__ fu, const float* __restrict__ fv,
                             const int* __restrict__ seg, const float* __restrict__ We,
                             float* __restrict__ e, int n) {
  int i = blockIdx.x * blockDim.x + threadIdx.x;
  if (i >= n) return;
  const float* up = fu + (size_t)i * 128;
  const float* vp = fv + (size_t)seg[i] * 128;
  float acc = 0.f;
#pragma unroll 8
  for (int d = 0; d < 128; ++d) acc += We[d] * sigm(up[d] + vp[d]);
  e[i] = acc;
}

__global__ void k_seg_max(const float* __restrict__ e, const int* __restrict__ seg,
                          unsigned* __restrict__ mx, int n) {
  int i = blockIdx.x * blockDim.x + threadIdx.x;
  if (i >= n) return;
  atomicMax(&mx[seg[i]], fenc(e[i]));
}

__global__ void k_seg_expsum(const float* __restrict__ e, const int* __restrict__ seg,
                             const unsigned* __restrict__ mx,
                             float* __restrict__ ex, float* __restrict__ sum, int n) {
  int i = blockIdx.x * blockDim.x + threadIdx.x;
  if (i >= n) return;
  float x = __expf(e[i] - fdec(mx[seg[i]]));
  ex[i] = x;
  atomicAdd(&sum[seg[i]], x);
}

__global__ void k_sgat_scatter(const float* __restrict__ ex, const float* __restrict__ sum,
                               const int* __restrict__ src, const int* __restrict__ dst,
                               const float* __restrict__ v, float* __restrict__ rst, int ne) {
  int i = blockIdx.x * blockDim.x + threadIdx.x;
  if (i >= ne * 128) return;
  int e = i >> 7, d = i & 127;
  int dd = dst[e];
  float a = ex[e] / sum[dd];
  atomicAdd(&rst[(size_t)dd * 128 + d], v[(size_t)src[e] * 128 + d] * a);
}

__global__ void k_readout_scatter(const float* __restrict__ xf, const float* __restrict__ ex,
                                  const float* __restrict__ sum, const int* __restrict__ seg,
                                  float* __restrict__ rst, int n) {
  int i = blockIdx.x * blockDim.x + threadIdx.x;
  if (i >= n * 768) return;
  int nd = i / 768, d = i - nd * 768;
  int s = seg[nd];
  float a = ex[nd] / sum[s];
  atomicAdd(&rst[(size_t)s * 768 + d], xf[i] * a);
}

__global__ void k_f2bf(const float* __restrict__ s, unsigned short* __restrict__ d, int n) {
  int i = blockIdx.x * blockDim.x + threadIdx.x;
  if (i < n) d[i] = f2bf(s[i]);
}

// ---------------------------------------------------------------------------
// Host orchestration
// ---------------------------------------------------------------------------
struct WDesc { int k, n, idx; };

extern "C" void kernel_launch(void* const* d_in, const int* in_sizes, int n_in,
                              void* d_out, int out_size, void* d_ws, size_t ws_size,
                              hipStream_t stream) {
  (void)in_sizes; (void)n_in; (void)out_size; (void)ws_size;
  auto F = [&](int i) { return (const float*)d_in[i]; };
  auto I = [&](int i) { return (const int*)d_in[i]; };
  auto g1 = [](long n) { return dim3((unsigned)((n + 255) / 256)); };

  // ---- workspace layout (bytes) ----
  char* ws = (char*)d_ws;
  const size_t OFF_TNORM = 0;                                   // V*D f32
  const size_t OFF_TBF   = OFF_TNORM + (size_t)V_ * D_ * 4;     // V*D bf16
  const size_t OFF_WA    = OFF_TBF + (size_t)V_ * D_ * 2;       // bf16 weight arena 4MB
  const size_t OFF_FEAT  = OFF_WA + (4u << 20);                 // N*768 f32
  const size_t OFF_XF    = OFF_FEAT + (size_t)N_ * 768 * 4;     // N*768 f32
  const size_t OFF_XBF   = OFF_XF + (size_t)N_ * 768 * 4;       // N*768 bf16
  const size_t OFF_GI    = OFF_XBF + (size_t)N_ * 768 * 2;      // N*1152 f32
  const size_t OFF_GH    = OFF_GI + (size_t)N_ * 1152 * 4;      // N*1152 f32
  const size_t OFF_H     = OFF_GH + (size_t)N_ * 1152 * 4;      // N*384 f32
  const size_t OFF_HBF   = OFF_H + (size_t)N_ * 384 * 4;        // N*384 bf16
  const size_t OFF_MISC  = OFF_HBF + (size_t)N_ * 384 * 2;      // small stuff

  float*          tnorm = (float*)(ws + OFF_TNORM);
  unsigned short* tbf   = (unsigned short*)(ws + OFF_TBF);
  float*          feat  = (float*)(ws + OFF_FEAT);
  float*          xf    = (float*)(ws + OFF_XF);
  unsigned short* xbf   = (unsigned short*)(ws + OFF_XBF);
  float*          GIf   = (float*)(ws + OFF_GI);
  float*          GHf   = (float*)(ws + OFF_GH);
  float*          Hf    = (float*)(ws + OFF_H);
  unsigned short* Hbf   = (unsigned short*)(ws + OFF_HBF);

  // SGAT / readout buffers carved from GI region (free when SGAT/readout run)
  float* Qf   = GIf;                                  // N*128 (also readout fu)
  float* Kf   = GIf + (size_t)N_ * 128;
  float* Vf   = GIf + (size_t)N_ * 128 * 2;
  float* RSTN = GIf + (size_t)N_ * 128 * 3;           // N*128
  // B-sized buffers carved from GH region (free at readout)
  char* p = ws + OFF_GH;
  auto carve = [&](size_t bytes) { char* r = p; p += (bytes + 255) & ~(size_t)255; return r; };
  float*          XL     = (float*)carve((size_t)B_ * 768 * 4);
  unsigned short* XLBF   = (unsigned short*)carve((size_t)B_ * 768 * 2);
  float*          FVB    = (float*)carve((size_t)B_ * 128 * 4);
  float*          RSTB   = (float*)carve((size_t)B_ * 768 * 4);
  unsigned short* RSTBBF = (unsigned short*)carve((size_t)B_ * 768 * 2);
  float*          SR     = (float*)carve((size_t)B_ * 896 * 4);
  float*          SRN    = (float*)carve((size_t)B_ * 896 * 4);
  unsigned short* SRNBF  = (unsigned short*)carve((size_t)B_ * 896 * 2);
  float*          SR2    = (float*)carve((size_t)B_ * 128 * 4);
  unsigned short* SR2BF  = (unsigned short*)carve((size_t)B_ * 128 * 2);
  float*          TMPB   = (float*)carve((size_t)B_ * 128 * 4);
  // misc region
  char* q = ws + OFF_MISC;
  auto carveM = [&](size_t bytes) { char* r = q; q += (bytes + 255) & ~(size_t)255; return r; };
  float*    Ef   = (float*)carveM((size_t)E_ * 4);
  float*    EXf  = (float*)carveM((size_t)E_ * 4);
  unsigned* NMAX = (unsigned*)carveM((size_t)N_ * 4);
  float*    NSUM = (float*)carveM((size_t)N_ * 4);
  float*    MU   = (float*)carveM(1024 * 4);
  float*    RS   = (float*)carveM(1024 * 4);

  // ---- prep: embedding max-norm + bf16; weight transpose/convert ----
  hipMemsetAsync(feat, 0, (size_t)N_ * 768 * 4, stream);        // sem cols stay 0
  k_norm_emb<<<dim3(V_), 128, 0, stream>>>(F(2), tnorm, tbf);

  // weight arena: WT[i] = transposed bf16 weight i
  static const WDesc wdesc[18] = {
    {128, 384, 5}, {128, 384, 4}, {128, 128, 7}, {128, 128, 6},   // L0 Wih,Whh,Wself,Wneigh
    {256, 128, 15}, {256, 128, 14}, {256, 128, 16},               // L1 Wq,Wk,Wv
    {384, 1152, 22}, {384, 1152, 21}, {384, 128, 24}, {384, 128, 23}, // L2
    {512, 128, 32}, {512, 128, 31}, {512, 128, 33},               // L3 Wq,Wk,Wv
    {768, 128, 40}, {768, 128, 41}, {768, 128, 39},               // RO Wu,Wv,Wout
    {896, 128, 3},                                                // fc_sr
  };
  unsigned short* WT[18];
  {
    size_t off = 0;
    for (int i = 0; i < 18; ++i) {
      WT[i] = (unsigned short*)(ws + OFF_WA) + off;
      long n = (long)wdesc[i].k * wdesc[i].n;
      k_convertT<<<g1(n), 256, 0, stream>>>(F(wdesc[i].idx), WT[i], wdesc[i].k, wdesc[i].n);
      off += ((size_t)n + 127) & ~(size_t)127;                   // 256B aligned
    }
  }

  auto gemm = [&](const unsigned short* A, int lda, const unsigned short* BT, int ldb,
                  float* C, int ldc, int M, int Nc, int K, const float* bias, int flags) {
    int tilesN = Nc / 16;
    int waveCols = (tilesN + 1) / 2;                 // 2 N-tiles per wave
    dim3 grid((waveCols + 3) / 4, M / 32);           // 4 waves/block, 2 M-tiles per wave
    k_gemm_bf16<<<grid, 128, 0, stream>>>(A, lda, BT, ldb, C, ldc, M, Nc, K, bias, flags);
  };
  auto bn = [&](const float* src, int stride, int rows, int C,
                const float* g_, const float* b_, float* of, unsigned short* ob) {
    k_bn_stats<<<dim3(C), 256, 0, stream>>>(src, stride, rows, MU, RS);
    k_bn_apply<<<g1((long)rows * C), 256, 0, stream>>>(src, stride, MU, RS, g_, b_, of, ob, C, rows);
  };

  // feat0 = t_norm[iid] into columns [512:640)
  k_gather_rows<<<g1((long)N_ * 128), 256, 0, stream>>>(
      tnorm, 128, I(47), feat + 512, (unsigned short*)nullptr, 768, 128, N_);

  auto eopa = [&](int din, int inCol, int outCol, int wIh, int wHh, int wSelf, int wNeigh,
                  int iBih, int iBhh, int iBng, int iBnb, int iA) {
    int tri = 3 * din;
    bn(feat + inCol, 768, N_, din, F(iBng), F(iBnb), xf, xbf);
    gemm(xbf, din, WT[wIh], din, GIf, tri, N_, tri, din, F(iBih), 1);  // Gi = x@Wih+bih
    hipMemsetAsync(Hf, 0, (size_t)N_ * din * 4, stream);
    hipMemsetAsync(Hbf, 0, (size_t)N_ * din * 2, stream);
    for (int t = 0; t < T_; ++t) {
      gemm(Hbf, din, WT[wHh], din, GHf, tri, N_, tri, din, F(iBhh), 1); // Gh = h@Whh+bhh
      k_gru_step<<<g1((long)N_ * din), 256, 0, stream>>>(
          GIf, GHf, I(48), (const unsigned char*)d_in[49], t, Hf, Hbf, din, N_);
    }
    gemm(xbf, din, WT[wSelf], din, feat + outCol, 768, N_, 128, din, nullptr, 0);
    gemm(Hbf, din, WT[wNeigh], din, feat + outCol, 768, N_, 128, din, nullptr, 2);
    k_prelu_copy<<<g1((long)N_ * 128), 256, 0, stream>>>(
        feat + outCol, 768, feat + outCol, 768, F(iA), N_);
  };

  auto sgat = [&](int din, int inCol, int outCol, int wQ, int wK, int wV,
                  int iWe, int iBq, int iBng, int iBnb, int iA) {
    bn(feat + inCol, 768, N_, din, F(iBng), F(iBnb), xf, xbf);
    gemm(xbf, din, WT[wQ], din, Qf, 128, N_, 128, din, F(iBq), 1);
    gemm(xbf, din, WT[wK], din, Kf, 128, N_, 128, din, nullptr, 0);
    gemm(xbf, din, WT[wV], din, Vf, 128, N_, 128, din, nullptr, 0);
    k_edge_score<<<g1(E_), 256, 0, stream>>>(Qf, Kf, F(iWe), I(50), I(51), Ef, E_);
    hipMemsetAsync(NMAX, 0, (size_t)N_ * 4, stream);
    hipMemsetAsync(NSUM, 0, (size_t)N_ * 4, stream);
    hipMemsetAsync(RSTN, 0, (size_t)N_ * 128 * 4, stream);
    k_seg_max<<<g1(E_), 256, 0, stream>>>(Ef, I(51), NMAX, E_);
    k_seg_expsum<<<g1(E_), 256, 0, stream>>>(Ef, I(51), NMAX, EXf, NSUM, E_);
    k_sgat_scatter<<<g1((long)E_ * 128), 256, 0, stream>>>(EXf, NSUM, I(50), I(51), Vf, RSTN, E_);
    k_prelu_copy<<<g1((long)N_ * 128), 256, 0, stream>>>(RSTN, 128, feat + outCol, 768, F(iA), N_);
  };

  // ---- 4 layers (concat layout: out3@0, out2@128, out1@256, out0@384, feat0@512, sem@640) ----
  eopa(128, 512, 384, 0, 1, 2, 3, 10, 9, 12, 11, 8);
  sgat(256, 384, 256, 4, 5, 6, 13, 20, 19, 18, 17);
  eopa(384, 256, 128, 7, 8, 9, 10, 27, 26, 29, 28, 25);
  sgat(512, 128, 0, 11, 12, 13, 30, 37, 36, 35, 34);

  // ---- attention readout over [N,768] ----
  bn(feat, 768, N_, 768, F(44), F(43), xf, xbf);
  gemm(xbf, 768, WT[14], 768, Qf /*fu*/, 128, N_, 128, 768, nullptr, 0);
  k_gather_rows<<<g1((long)B_ * 768), 256, 0, stream>>>(xf, 768, I(53), XL, XLBF, 768, 768, B_);
  gemm(XLBF, 768, WT[15], 768, FVB, 128, B_, 128, 768, F(45), 1);
  k_node_score<<<g1(N_), 256, 0, stream>>>(Qf, FVB, I(52), F(38), Ef, N_);
  hipMemsetAsync(NMAX, 0, (size_t)B_ * 4, stream);
  hipMemsetAsync(NSUM, 0, (size_t)B_ * 4, stream);
  hipMemsetAsync(RSTB, 0, (size_t)B_ * 768 * 4, stream);
  k_seg_max<<<g1(N_), 256, 0, stream>>>(Ef, I(52), NMAX, N_);
  k_seg_expsum<<<g1(N_), 256, 0, stream>>>(Ef, I(52), NMAX, EXf, NSUM, N_);
  k_readout_scatter<<<g1((long)N_ * 768), 256, 0, stream>>>(xf, EXf, NSUM, I(52), RSTB, N_);
  k_f2bf<<<g1((long)B_ * 768), 256, 0, stream>>>(RSTB, RSTBBF, B_ * 768);
  gemm(RSTBBF, 768, WT[16], 768, TMPB, 128, B_, 128, 768, nullptr, 0);
  k_prelu_copy<<<g1((long)B_ * 128), 256, 0, stream>>>(TMPB, 128, SR + 768, 896, F(42), B_);

  // ---- sr = concat(feat[last], sr_g); BN; fc_sr; scores = sr @ t^T ----
  k_gather_rows<<<g1((long)B_ * 768), 256, 0, stream>>>(
      feat, 768, I(53), SR, (unsigned short*)nullptr, 896, 768, B_);
  bn(SR, 896, B_, 896, F(1), F(0), SRN, SRNBF);
  gemm(SRNBF, 896, WT[17], 896, SR2, 128, B_, 128, 896, nullptr, 0);
  k_f2bf<<<g1((long)B_ * 128), 256, 0, stream>>>(SR2, SR2BF, B_ * 128);
  gemm(SR2BF, 128, tbf, 128, (float*)d_out, V_, B_, V_, 128, nullptr, 0);
}
